// TransformerBlockWithMoE_85590108275213
// MI455X (gfx1250) — matrix operand order repaired and checked by reading, not compiled, not run on hardware
//
#include <hip/hip_runtime.h>

// ---------------------------------------------------------------------------
// MoE transformer block for MI455X (gfx1250, wave32, WMMA bf16 16x16x32).
// Sparse routing (top-2 of 8) == reference's dense-expert + sparse-gate math.
// bf16 weights (128MB) fit in the 192MB L2. GEMM inner loop: LDS tiles are
// pre-packed into WMMA fragment layout so each fragment is 2x ds_load_b128
// (no VALU packing -> no WMMA->VALU WAR NOPs). Rows padded to 20 dwords for
// bank-conflict-free strided b128 reads.
// ---------------------------------------------------------------------------

#define N_TOK 8192
#define D_DIM 1024
#define H_DIM 4096
#define E_NUM 8
#define K_TOP 2
#define CAP   16640            // padded slot capacity (multiple of 32)
#define BROWS 32               // rows per GEMM block (two 16-row WMMA halves)
#define RT_COUNT (CAP / BROWS) // 520 row tiles

typedef __attribute__((ext_vector_type(16))) __bf16 v16bf;
typedef __attribute__((ext_vector_type(8)))  float  v8f;

union FragAB {
    v16bf v;
    unsigned int u[8];
};

__device__ __forceinline__ unsigned short f2bf(float f) {
    unsigned int u = __float_as_uint(f);
    unsigned int r = u + 0x7FFFu + ((u >> 16) & 1u);   // round to nearest even
    return (unsigned short)(r >> 16);
}

// ------------------------------ casts --------------------------------------
__global__ __launch_bounds__(256) void cast_f32_bf16_kernel(
    const float* __restrict__ src, unsigned short* __restrict__ dst, size_t n) {
    size_t i = (size_t)blockIdx.x * blockDim.x + threadIdx.x;
    size_t stride = (size_t)gridDim.x * blockDim.x;
    for (; i < n; i += stride) dst[i] = f2bf(src[i]);
}

// ------------------------------ gating -------------------------------------
__global__ __launch_bounds__(256) void gating_kernel(
    const float* __restrict__ x, const float* __restrict__ Wg,
    const float* __restrict__ bg,
    int* __restrict__ topi, float* __restrict__ topw) {
    const int lane = threadIdx.x & 31;
    const int wave = threadIdx.x >> 5;
    const int n = blockIdx.x * 8 + wave;
    const float* xr = x + (size_t)n * D_DIM;

    float acc[E_NUM] = {0.f, 0.f, 0.f, 0.f, 0.f, 0.f, 0.f, 0.f};
    for (int d = lane * 32; d < lane * 32 + 32; ++d) {
        float xv = xr[d];
        const float* wr = Wg + (size_t)d * E_NUM;
#pragma unroll
        for (int e = 0; e < E_NUM; ++e) acc[e] += xv * wr[e];
    }
#pragma unroll
    for (int e = 0; e < E_NUM; ++e) {
        float v = acc[e];
#pragma unroll
        for (int m = 16; m >= 1; m >>= 1) v += __shfl_xor(v, m, 32);
        acc[e] = v + bg[e];
    }
    if (lane == 0) {
        int i0 = 0; float v0 = acc[0];
#pragma unroll
        for (int e = 1; e < E_NUM; ++e) if (acc[e] > v0) { v0 = acc[e]; i0 = e; }
        int i1 = (i0 == 0) ? 1 : 0; float v1 = acc[i1];
#pragma unroll
        for (int e = 0; e < E_NUM; ++e)
            if (e != i0 && acc[e] > v1) { v1 = acc[e]; i1 = e; }
        float e1 = __expf(v1 - v0);          // v0 >= v1
        float inv = 1.f / (1.f + e1);
        topi[n * 2 + 0] = i0; topi[n * 2 + 1] = i1;
        topw[n * 2 + 0] = inv; topw[n * 2 + 1] = e1 * inv;
    }
}

// ------------------------------ prep (1 block) -----------------------------
__global__ __launch_bounds__(256) void prep_kernel(
    const int* __restrict__ topi, const float* __restrict__ topw,
    int* __restrict__ gmeta,            // [0..8]=offsets, [9..16]=cursor
    int* __restrict__ token_of, int* __restrict__ rt_expert,
    float* __restrict__ lb_out) {
    __shared__ int   cntArr[256 * E_NUM];
    __shared__ float diArr[256 * E_NUM];
    __shared__ int   soffs[E_NUM + 1];
    __shared__ int   sCnt[E_NUM];
    __shared__ float sDi[E_NUM];
    const int tid = threadIdx.x;

    int   lc[E_NUM] = {0, 0, 0, 0, 0, 0, 0, 0};
    float ls[E_NUM] = {0.f, 0.f, 0.f, 0.f, 0.f, 0.f, 0.f, 0.f};
    for (int i = tid; i < N_TOK * K_TOP; i += 256) {
        int e = topi[i];
        lc[e] += 1;
        ls[e] += topw[i];
    }
#pragma unroll
    for (int e = 0; e < E_NUM; ++e) {
        cntArr[tid * E_NUM + e] = lc[e];
        diArr[tid * E_NUM + e] = ls[e];
    }
    __syncthreads();
    if (tid < E_NUM) {
        int c = 0; float d = 0.f;
        for (int t = 0; t < 256; ++t) { c += cntArr[t * E_NUM + tid]; d += diArr[t * E_NUM + tid]; }
        sCnt[tid] = c; sDi[tid] = d;
        gmeta[9 + tid] = 0;                 // cursors
    }
    __syncthreads();
    if (tid == 0) {
        int off = 0; float lb = 0.f;
        for (int e = 0; e < E_NUM; ++e) {
            soffs[e] = off;
            off += (sCnt[e] + (BROWS - 1)) & ~(BROWS - 1);   // 32-align segments
            float Di = sDi[e] / (float)N_TOK;
            lb += Di * logf(Di + 1e-8f);
        }
        soffs[E_NUM] = CAP;
        for (int e = 0; e <= E_NUM; ++e) gmeta[e] = soffs[e];
        *lb_out = lb;
    }
    __syncthreads();
    for (int i = tid; i < CAP; i += 256) token_of[i] = -1;
    for (int rt = tid; rt < RT_COUNT; rt += 256) {
        int r0 = rt * BROWS, e = 0;
        for (int j = E_NUM - 1; j >= 1; --j)
            if (r0 >= soffs[j]) { e = j; break; }
        rt_expert[rt] = e;
    }
}

// ------------------------------ assign -------------------------------------
__global__ __launch_bounds__(256) void assign_kernel(
    const int* __restrict__ topi, const float* __restrict__ topw,
    int* __restrict__ gmeta, int* __restrict__ token_of,
    float* __restrict__ w_of, int* __restrict__ slot_of) {
    int i = blockIdx.x * 256 + threadIdx.x;       // < N_TOK*K_TOP
    int e = topi[i];
    int pos = atomicAdd(&gmeta[9 + e], 1);
    int slot = gmeta[e] + pos;
    token_of[slot] = i >> 1;
    w_of[slot] = topw[i];
    slot_of[i] = slot;
}

// ------------------------------ GEMM 1 -------------------------------------
// h[slot, :] = relu( x[token_of[slot], :] @ W1[e] + b1[e] )   (bf16 out)
// block: 32 rows x 512 cols, 8 waves x 4 col tiles x 2 row halves, K-step 32
__global__ __launch_bounds__(256) void gemm1_kernel(
    const unsigned short* __restrict__ xb,     // [N, D] bf16
    const unsigned short* __restrict__ W1b,    // [E, D, H] bf16
    const float* __restrict__ b1,              // [E, H]
    const int* __restrict__ token_of,          // [CAP]
    const int* __restrict__ rt_expert,         // [RT_COUNT]
    unsigned short* __restrict__ hbuf) {       // [CAP, H] bf16
    // A tile 32x32 bf16: row-major, K-pairs contiguous (frag = 2x b128)
    __shared__ unsigned int As[BROWS][20];     // 2.5 KB (padded: conflict-free)
    __shared__ unsigned int Bp[512][20];       // 40 KB, Bp[c][kp]=B[2kp][c]|B[2kp+1][c]<<16
    __shared__ int sTok[BROWS];
    const int rt = blockIdx.y;
    const int c0 = blockIdx.x * 512;
    const int e = rt_expert[rt];
    const int tid = threadIdx.x;
    const int lane = tid & 31, wave = tid >> 5;
    const int g = lane >> 4, nl = lane & 15;
    const unsigned short* wbase = W1b + (size_t)e * D_DIM * H_DIM;

    if (tid < BROWS) sTok[tid] = token_of[rt * BROWS + tid];

    const v8f z = {0.f, 0.f, 0.f, 0.f, 0.f, 0.f, 0.f, 0.f};
    v8f acc[2][4] = {{z, z, z, z}, {z, z, z, z}};

    for (int kk = 0; kk < D_DIM; kk += 32) {
        __syncthreads();
        // stage A: 32 rows x 16 dwords, gathered by token (padded rows -> 0)
        for (int i = tid; i < 512; i += 256) {
            int r = i >> 4, cu = i & 15;
            int tok = sTok[r];
            unsigned int v = 0;
            if (tok >= 0) v = *(const unsigned int*)&xb[(size_t)tok * D_DIM + kk + cu * 2];
            As[r][cu] = v;
        }
        // stage B packed: 64 col-octets x 16 k-pairs
        for (int i = tid; i < 1024; i += 256) {
            int kp = i & 15;
            int c8 = (i >> 4) * 8;
            const unsigned short* r0 = &wbase[(size_t)(kk + 2 * kp) * H_DIM + c0 + c8];
            uint4 lo4 = *(const uint4*)r0;
            uint4 hi4 = *(const uint4*)(r0 + H_DIM);
            const unsigned short* lo = (const unsigned short*)&lo4;
            const unsigned short* hi = (const unsigned short*)&hi4;
#pragma unroll
            for (int j = 0; j < 8; ++j)
                Bp[c8 + j][kp] = (unsigned int)lo[j] | ((unsigned int)hi[j] << 16);
        }
        __syncthreads();

        FragAB fa[2];
#pragma unroll
        for (int h = 0; h < 2; ++h) {
            const unsigned int* ar = As[h * 16 + nl];
            *(uint4*)&fa[h].u[0] = *(const uint4*)&ar[4 * g];
            *(uint4*)&fa[h].u[4] = *(const uint4*)&ar[8 + 4 * g];
        }
#pragma unroll
        for (int t = 0; t < 4; ++t) {
            const unsigned int* br = Bp[(wave * 4 + t) * 16 + nl];
            FragAB fb;
            *(uint4*)&fb.u[0] = *(const uint4*)&br[4 * g];
            *(uint4*)&fb.u[4] = *(const uint4*)&br[8 + 4 * g];
#pragma unroll
            for (int h = 0; h < 2; ++h)
                acc[h][t] = __builtin_amdgcn_wmma_f32_16x16x32_bf16(
                    false, fa[h].v, false, fb.v, (short)0, acc[h][t], false, false);
        }
    }
    // epilogue: +bias, relu, pack bf16
#pragma unroll
    for (int t = 0; t < 4; ++t) {
        int col = c0 + (wave * 4 + t) * 16 + nl;
        float bias = b1[e * H_DIM + col];
#pragma unroll
        for (int h = 0; h < 2; ++h) {
#pragma unroll
            for (int v = 0; v < 8; ++v) {
                int slot = rt * BROWS + h * 16 + v + g * 8;
                float val = fmaxf(acc[h][t][v] + bias, 0.f);
                hbuf[(size_t)slot * H_DIM + col] = f2bf(val);
            }
        }
    }
}

// ------------------------------ GEMM 2 -------------------------------------
// eo[slot, :] = gate_w[slot] * ( h[slot,:] @ W2[e] + b2[e] )   (f32 out)
__global__ __launch_bounds__(256) void gemm2_kernel(
    const unsigned short* __restrict__ hbuf,   // [CAP, H] bf16
    const unsigned short* __restrict__ W2b,    // [E, H, D] bf16
    const float* __restrict__ b2,              // [E, D]
    const float* __restrict__ w_of,            // [CAP]
    const int* __restrict__ rt_expert,
    float* __restrict__ eo) {                  // [CAP, D] f32
    __shared__ unsigned int As[BROWS][20];
    __shared__ unsigned int Bp[512][20];
    const int rt = blockIdx.y;
    const int c0 = blockIdx.x * 512;
    const int e = rt_expert[rt];
    const int tid = threadIdx.x;
    const int lane = tid & 31, wave = tid >> 5;
    const int g = lane >> 4, nl = lane & 15;
    const unsigned short* wbase = W2b + (size_t)e * H_DIM * D_DIM;
    const unsigned short* hbase = hbuf + (size_t)rt * BROWS * H_DIM;

    const v8f z = {0.f, 0.f, 0.f, 0.f, 0.f, 0.f, 0.f, 0.f};
    v8f acc[2][4] = {{z, z, z, z}, {z, z, z, z}};

    for (int kk = 0; kk < H_DIM; kk += 32) {
        __syncthreads();
        for (int i = tid; i < 512; i += 256) {
            int r = i >> 4, cu = i & 15;
            As[r][cu] = *(const unsigned int*)&hbase[(size_t)r * H_DIM + kk + cu * 2];
        }
        for (int i = tid; i < 1024; i += 256) {
            int kp = i & 15;
            int c8 = (i >> 4) * 8;
            const unsigned short* r0 = &wbase[(size_t)(kk + 2 * kp) * D_DIM + c0 + c8];
            uint4 lo4 = *(const uint4*)r0;
            uint4 hi4 = *(const uint4*)(r0 + D_DIM);
            const unsigned short* lo = (const unsigned short*)&lo4;
            const unsigned short* hi = (const unsigned short*)&hi4;
#pragma unroll
            for (int j = 0; j < 8; ++j)
                Bp[c8 + j][kp] = (unsigned int)lo[j] | ((unsigned int)hi[j] << 16);
        }
        __syncthreads();

        FragAB fa[2];
#pragma unroll
        for (int h = 0; h < 2; ++h) {
            const unsigned int* ar = As[h * 16 + nl];
            *(uint4*)&fa[h].u[0] = *(const uint4*)&ar[4 * g];
            *(uint4*)&fa[h].u[4] = *(const uint4*)&ar[8 + 4 * g];
        }
#pragma unroll
        for (int t = 0; t < 4; ++t) {
            const unsigned int* br = Bp[(wave * 4 + t) * 16 + nl];
            FragAB fb;
            *(uint4*)&fb.u[0] = *(const uint4*)&br[4 * g];
            *(uint4*)&fb.u[4] = *(const uint4*)&br[8 + 4 * g];
#pragma unroll
            for (int h = 0; h < 2; ++h)
                acc[h][t] = __builtin_amdgcn_wmma_f32_16x16x32_bf16(
                    false, fa[h].v, false, fb.v, (short)0, acc[h][t], false, false);
        }
    }
#pragma unroll
    for (int t = 0; t < 4; ++t) {
        int col = c0 + (wave * 4 + t) * 16 + nl;
        float bias = b2[e * D_DIM + col];
#pragma unroll
        for (int h = 0; h < 2; ++h) {
#pragma unroll
            for (int v = 0; v < 8; ++v) {
                int slot = rt * BROWS + h * 16 + v + g * 8;
                eo[(size_t)slot * D_DIM + col] = (acc[h][t][v] + bias) * w_of[slot];
            }
        }
    }
}

// ------------------------ residual + LayerNorm gather ----------------------
__global__ __launch_bounds__(256) void fuse_ln_kernel(
    const float* __restrict__ x, const float* __restrict__ eo,
    const int* __restrict__ slot_of,
    const float* __restrict__ gamma, const float* __restrict__ beta,
    float* __restrict__ out) {
    __shared__ float rs[8], rq[8];
    __shared__ float sMu, sInv;
    const int n = blockIdx.x, tid = threadIdx.x;
    const int s0 = slot_of[n * 2], s1 = slot_of[n * 2 + 1];
    const float* xr = x + (size_t)n * D_DIM;
    const float* e0 = eo + (size_t)s0 * D_DIM;
    const float* e1 = eo + (size_t)s1 * D_DIM;

    float yv[4], s = 0.f, q = 0.f;
#pragma unroll
    for (int j = 0; j < 4; ++j) {
        int d = tid + j * 256;
        float y = xr[d] + e0[d] + e1[d];
        yv[j] = y; s += y; q += y * y;
    }
#pragma unroll
    for (int m = 16; m >= 1; m >>= 1) {
        s += __shfl_xor(s, m, 32);
        q += __shfl_xor(q, m, 32);
    }
    if ((tid & 31) == 0) { rs[tid >> 5] = s; rq[tid >> 5] = q; }
    __syncthreads();
    if (tid == 0) {
        float ts = 0.f, tq = 0.f;
        for (int i = 0; i < 8; ++i) { ts += rs[i]; tq += rq[i]; }
        float mu = ts / (float)D_DIM;
        float var = tq / (float)D_DIM - mu * mu;
        sMu = mu; sInv = rsqrtf(var + 1e-5f);
    }
    __syncthreads();
    const float mu = sMu, inv = sInv;
    float* orow = out + (size_t)n * D_DIM;
#pragma unroll
    for (int j = 0; j < 4; ++j) {
        int d = tid + j * 256;
        orow[d] = (yv[j] - mu) * inv * gamma[d] + beta[d];
    }
}

// ---------------------------------------------------------------------------
extern "C" void kernel_launch(void* const* d_in, const int* in_sizes, int n_in,
                              void* d_out, int out_size, void* d_ws, size_t ws_size,
                              hipStream_t stream) {
    (void)in_sizes; (void)n_in; (void)out_size; (void)ws_size;
    const float* x     = (const float*)d_in[0];
    const float* Wg    = (const float*)d_in[1];
    const float* bg    = (const float*)d_in[2];
    const float* W1    = (const float*)d_in[3];
    const float* b1    = (const float*)d_in[4];
    const float* W2    = (const float*)d_in[5];
    const float* b2    = (const float*)d_in[6];
    const float* gamma = (const float*)d_in[7];
    const float* beta  = (const float*)d_in[8];
    float* out = (float*)d_out;

    char* ws = (char*)d_ws;
    size_t off = 0;
    auto alloc = [&](size_t bytes) -> void* {
        void* p = ws + off;
        off += (bytes + 255) & ~(size_t)255;
        return p;
    };
    unsigned short* xb       = (unsigned short*)alloc((size_t)N_TOK * D_DIM * 2);
    unsigned short* W1b      = (unsigned short*)alloc((size_t)E_NUM * D_DIM * H_DIM * 2);
    unsigned short* W2b      = (unsigned short*)alloc((size_t)E_NUM * H_DIM * D_DIM * 2);
    unsigned short* hbuf     = (unsigned short*)alloc((size_t)CAP * H_DIM * 2);
    float*          eo       = (float*)alloc((size_t)CAP * D_DIM * 4);
    int*            topi     = (int*)alloc((size_t)N_TOK * K_TOP * 4);
    float*          topw     = (float*)alloc((size_t)N_TOK * K_TOP * 4);
    int*            slot_of  = (int*)alloc((size_t)N_TOK * K_TOP * 4);
    int*            token_of = (int*)alloc((size_t)CAP * 4);
    float*          w_of     = (float*)alloc((size_t)CAP * 4);
    int*            rt_exp   = (int*)alloc((size_t)RT_COUNT * 4);
    int*            gmeta    = (int*)alloc(32 * 4);

    // 1) casts to bf16
    cast_f32_bf16_kernel<<<4096, 256, 0, stream>>>(x, xb, (size_t)N_TOK * D_DIM);
    cast_f32_bf16_kernel<<<8192, 256, 0, stream>>>(W1, W1b, (size_t)E_NUM * D_DIM * H_DIM);
    cast_f32_bf16_kernel<<<8192, 256, 0, stream>>>(W2, W2b, (size_t)E_NUM * H_DIM * D_DIM);
    // 2) gating (one wave32 per token)
    gating_kernel<<<N_TOK / 8, 256, 0, stream>>>(x, Wg, bg, topi, topw);
    // 3) offsets + lb_loss + maps (deterministic single-block reductions)
    prep_kernel<<<1, 256, 0, stream>>>(topi, topw, gmeta, token_of, rt_exp,
                                       out + (size_t)N_TOK * D_DIM);
    // 4) token -> slot scatter
    assign_kernel<<<(N_TOK * K_TOP) / 256, 256, 0, stream>>>(topi, topw, gmeta,
                                                             token_of, w_of, slot_of);
    // 5) grouped expert GEMMs (WMMA bf16, f32 accum)
    gemm1_kernel<<<dim3(H_DIM / 512, RT_COUNT), 256, 0, stream>>>(
        xb, W1b, b1, token_of, rt_exp, hbuf);
    gemm2_kernel<<<dim3(D_DIM / 512, RT_COUNT), 256, 0, stream>>>(
        hbuf, W2b, b2, w_of, rt_exp, eo);
    // 6) residual + LayerNorm gather
    fuse_ln_kernel<<<N_TOK, 256, 0, stream>>>(x, eo, slot_of, gamma, beta, out);
}
